// SelfAM_39676907888323
// MI455X (gfx1250) — matrix operand order repaired and checked by reading, not compiled, not run on hardware
//
#include <hip/hip_runtime.h>
#include <hip/hip_bf16.h>

// ---------------------------------------------------------------------------
// Self-attention-style block for MI455X (gfx1250, wave32, WMMA)
//   V/K/Q = W*x + b ; G = V K^T ; A = sigmoid(G) Q * scale ; out = Wo A + bo + x
// All GEMMs run through v_wmma_f32_16x16x32_bf16 (f32 accumulate).
//
// Round-3 changes:
//  * Kernels 2-4 use 32x32 wave tiles (2 A frags x 2 B frags -> 4 WMMAs):
//    halves LDS fragment traffic per matrix op (2 frag loads per wmma -> 1).
//    Block tile 64(M) x 128(N), 8 waves arranged 2x4.
//  * Tile fills templated over row count (64/128), still fully unrolled.
// ---------------------------------------------------------------------------

typedef __attribute__((ext_vector_type(16))) __bf16 v16bf;
typedef __attribute__((ext_vector_type(8)))  float  v8f;
typedef __attribute__((ext_vector_type(4)))  __bf16 bf16x4;

#define B_    32
#define CIN_  256
#define CI_   128
#define HW_   4096

// Tile row stride in bf16 units: rows of 32 K-elements padded to 36
// (72B rows: 8B-aligned vector stores, 18-dword stride -> conflict-free
// half-wave fragment reads since 18*m mod 64 is distinct for m=0..15).
#define LDA_   36
#define TILE64_  (64 * LDA_)
#define TILE128_ (128 * LDA_)

// ---------------------------------------------------------------------------
// WMMA wrapper
// ---------------------------------------------------------------------------
static __device__ __forceinline__ v8f wmma_bf16(v16bf a, v16bf b, v8f c) {
  // 8 args: (neg_a, A, neg_b, B, c_mod, C, reuse_a, reuse_b)
  return __builtin_amdgcn_wmma_f32_16x16x32_bf16(false, a, false, b, (short)0, c,
                                                 false, false);
}

// ---------------------------------------------------------------------------
// LDS tile fills (256 threads, statically unrolled).  Tiles are
// [ROWS][32 K] with row stride LDA_, ROWS = 64 or 128.
// ---------------------------------------------------------------------------
template <int ROWS>
static __device__ __forceinline__ void fill_rows_f32(__bf16* lds, const float* src,
                                                     int srcld) {
#pragma unroll
  for (int rep = 0; rep < ROWS / 32; ++rep) {
    int idx = threadIdx.x + rep * 256;          // ROWS*8 float4 groups
    int m  = idx >> 3;
    int k4 = (idx & 7) << 2;
    float4 v = *(const float4*)(src + (size_t)m * srcld + k4);
    __bf16* d = lds + m * LDA_ + k4;
    d[0] = (__bf16)v.x; d[1] = (__bf16)v.y; d[2] = (__bf16)v.z; d[3] = (__bf16)v.w;
  }
}

template <int ROWS>
static __device__ __forceinline__ void fill_rows_bf(__bf16* lds, const __bf16* src,
                                                    int srcld) {
#pragma unroll
  for (int rep = 0; rep < ROWS / 32; ++rep) {
    int idx = threadIdx.x + rep * 256;
    int m  = idx >> 3;
    int k4 = (idx & 7) << 2;
    bf16x4 v = *(const bf16x4*)(src + (size_t)m * srcld + k4);
    *(bf16x4*)(lds + m * LDA_ + k4) = v;        // 8B-aligned (LDA_=36)
  }
}

// Transposing fills: source is [k=32][n=NROWS] row-major (contiguous in n, so
// global loads stay coalesced); LDS tile becomes [n=NROWS][k=32].
template <int NROWS>
static __device__ __forceinline__ void fill_BT_f32(__bf16* lds, const float* src,
                                                   int srcld) {
#pragma unroll
  for (int rep = 0; rep < NROWS / 32; ++rep) {
    int idx = threadIdx.x + rep * 256;
    int k  = idx / (NROWS / 4);
    int n4 = (idx % (NROWS / 4)) << 2;
    float4 v = *(const float4*)(src + (size_t)k * srcld + n4);
    lds[(n4 + 0) * LDA_ + k] = (__bf16)v.x;
    lds[(n4 + 1) * LDA_ + k] = (__bf16)v.y;
    lds[(n4 + 2) * LDA_ + k] = (__bf16)v.z;
    lds[(n4 + 3) * LDA_ + k] = (__bf16)v.w;
  }
}

template <int NROWS>
static __device__ __forceinline__ void fill_BT_bf(__bf16* lds, const __bf16* src,
                                                  int srcld) {
#pragma unroll
  for (int rep = 0; rep < NROWS / 32; ++rep) {
    int idx = threadIdx.x + rep * 256;
    int k  = idx / (NROWS / 4);
    int n4 = (idx % (NROWS / 4)) << 2;
    bf16x4 v = *(const bf16x4*)(src + (size_t)k * srcld + n4);
    lds[(n4 + 0) * LDA_ + k] = v.x;
    lds[(n4 + 1) * LDA_ + k] = v.y;
    lds[(n4 + 2) * LDA_ + k] = v.z;
    lds[(n4 + 3) * LDA_ + k] = v.w;
  }
}

// ---------------------------------------------------------------------------
// Fragment loaders (contiguous K order k = 16*half + s, wave32).  Both operands read
// rows of a [*][LDA_] tile -> contiguous per-lane dword pairs.
// ---------------------------------------------------------------------------
static __device__ __forceinline__ v16bf load_frag_A(const __bf16* lds, int m0) {
  int lane = threadIdx.x & 31;
  int half = lane >> 4;
  const __bf16* row = lds + (m0 + (lane & 15)) * LDA_;
  v16bf a;
#pragma unroll
  for (int j = 0; j < 4; ++j) {
    int k = half * 16 + 2 * j;      // VGPRs 0..3: K 16h+0..7
    a[2 * j]     = row[k];
    a[2 * j + 1] = row[k + 1];
    int k2 = 8 + k;                 // VGPRs 4..7: K 16h+8..15
    a[8 + 2 * j] = row[k2];
    a[9 + 2 * j] = row[k2 + 1];
  }
  return a;
}

static __device__ __forceinline__ v16bf load_frag_B(const __bf16* lds, int n0) {
  // B operand from transposed tile [n][k]: B[k][n] = tile[n][k]
  int lane = threadIdx.x & 31;
  int half = lane >> 4;
  const __bf16* row = lds + (n0 + (lane & 15)) * LDA_;
  v16bf b;
#pragma unroll
  for (int j = 0; j < 8; ++j) {
    int k = half * 16 + 2 * j;
    b[2 * j]     = row[k];
    b[2 * j + 1] = row[k + 1];
  }
  return b;
}

// ---------------------------------------------------------------------------
// Accumulator epilogues (f32 C/D 16x16 layout: VGPR r -> M = half*8 + r)
// ---------------------------------------------------------------------------
static __device__ __forceinline__ void store_bias_bf(__bf16* dst, v8f acc, int m0,
                                                     int n0, const float* bias) {
  int lane = threadIdx.x & 31;
  int n  = n0 + (lane & 15);
  int mb = m0 + ((lane >> 4) << 3);
#pragma unroll
  for (int r = 0; r < 8; ++r)
    dst[(size_t)(mb + r) * HW_ + n] = (__bf16)(acc[r] + bias[mb + r]);
}

static __device__ __forceinline__ void store_scaled_bf(__bf16* dst, v8f acc, int m0,
                                                       int n0, float scale) {
  int lane = threadIdx.x & 31;
  int n  = n0 + (lane & 15);
  int mb = m0 + ((lane >> 4) << 3);
#pragma unroll
  for (int r = 0; r < 8; ++r)
    dst[(size_t)(mb + r) * HW_ + n] = (__bf16)(acc[r] * scale);
}

static __device__ __forceinline__ void atomic_acc_f32(float* dst, v8f acc, int m0,
                                                      int n0) {
  int lane = threadIdx.x & 31;
  int n  = n0 + (lane & 15);
  int mb = m0 + ((lane >> 4) << 3);
#pragma unroll
  for (int r = 0; r < 8; ++r)
    atomicAdd(&dst[(size_t)(mb + r) * CI_ + n], acc[r]);
}

// ---------------------------------------------------------------------------
// Kernel 1: fused QKV projection.  64(ci) x 64(hw) block tile of all three
// outputs; the x tile is staged once and reused for V,K,Q.  Double-buffered.
// Wave tile 16x32: 5 fragment loads amortized over 6 WMMAs.
// ---------------------------------------------------------------------------
static __device__ __forceinline__ void qkv_fill(
    __bf16* w0, __bf16* w1, __bf16* w2, __bf16* xt,
    const float* Wv, const float* Wk, const float* Wq, const float* xb,
    int m0, int n0, int k0) {
  fill_rows_f32<64>(w0, Wv + (size_t)m0 * CIN_ + k0, CIN_);
  fill_rows_f32<64>(w1, Wk + (size_t)m0 * CIN_ + k0, CIN_);
  fill_rows_f32<64>(w2, Wq + (size_t)m0 * CIN_ + k0, CIN_);
  fill_BT_f32<64>(xt, xb + (size_t)k0 * HW_ + n0, HW_);
}

__global__ __launch_bounds__(256) void qkv_kernel(
    const float* __restrict__ x,
    const float* __restrict__ Wv, const float* __restrict__ bv,
    const float* __restrict__ Wk, const float* __restrict__ bk,
    const float* __restrict__ Wq, const float* __restrict__ bq,
    __bf16* __restrict__ V, __bf16* __restrict__ K, __bf16* __restrict__ Q) {
  __shared__ __align__(16) __bf16 sW[2][3][TILE64_];
  __shared__ __align__(16) __bf16 sX[2][TILE64_];

  int n0 = blockIdx.x * 64;
  int m0 = blockIdx.y * 64;
  int b  = blockIdx.z;
  const float* xb = x + (size_t)b * CIN_ * HW_;

  int wave = threadIdx.x >> 5;
  int mw = (wave >> 1) * 16;
  int nw = (wave & 1) * 32;

  v8f aV0 = {}, aV1 = {}, aK0 = {}, aK1 = {}, aQ0 = {}, aQ1 = {};

  qkv_fill(sW[0][0], sW[0][1], sW[0][2], sX[0], Wv, Wk, Wq, xb, m0, n0, 0);

  const int NT = CIN_ / 32;  // 8
  for (int kt = 0; kt < NT; ++kt) {
    __syncthreads();
    int cur = kt & 1;
    if (kt + 1 < NT)
      qkv_fill(sW[cur ^ 1][0], sW[cur ^ 1][1], sW[cur ^ 1][2], sX[cur ^ 1],
               Wv, Wk, Wq, xb, m0, n0, (kt + 1) * 32);

    v16bf av  = load_frag_A(sW[cur][0], mw);
    v16bf ak  = load_frag_A(sW[cur][1], mw);
    v16bf aq  = load_frag_A(sW[cur][2], mw);
    v16bf fb0 = load_frag_B(sX[cur], nw);
    v16bf fb1 = load_frag_B(sX[cur], nw + 16);

    aV0 = wmma_bf16(av, fb0, aV0);  aV1 = wmma_bf16(av, fb1, aV1);
    aK0 = wmma_bf16(ak, fb0, aK0);  aK1 = wmma_bf16(ak, fb1, aK1);
    aQ0 = wmma_bf16(aq, fb0, aQ0);  aQ1 = wmma_bf16(aq, fb1, aQ1);
  }

  size_t base = (size_t)b * CI_ * HW_;
  store_bias_bf(V + base, aV0, m0 + mw, n0 + nw, bv);
  store_bias_bf(V + base, aV1, m0 + mw, n0 + nw + 16, bv);
  store_bias_bf(K + base, aK0, m0 + mw, n0 + nw, bk);
  store_bias_bf(K + base, aK1, m0 + mw, n0 + nw + 16, bk);
  store_bias_bf(Q + base, aQ0, m0 + mw, n0 + nw, bq);
  store_bias_bf(Q + base, aQ1, m0 + mw, n0 + nw + 16, bq);
}

// ---------------------------------------------------------------------------
// Shared 32x32-wave-tile GEMM core for kernels 2-4.
// Block tile 64(M) x 128(N); waves arranged 2(M) x 4(N).
// ---------------------------------------------------------------------------
struct Acc4 { v8f a00, a01, a10, a11; };

static __device__ __forceinline__ void mma_step(Acc4& acc, const __bf16* tA,
                                                const __bf16* tB, int mw, int nw) {
  v16bf fa0 = load_frag_A(tA, mw);
  v16bf fa1 = load_frag_A(tA, mw + 16);
  v16bf fb0 = load_frag_B(tB, nw);
  v16bf fb1 = load_frag_B(tB, nw + 16);
  acc.a00 = wmma_bf16(fa0, fb0, acc.a00);
  acc.a01 = wmma_bf16(fa0, fb1, acc.a01);
  acc.a10 = wmma_bf16(fa1, fb0, acc.a10);
  acc.a11 = wmma_bf16(fa1, fb1, acc.a11);
}

// ---------------------------------------------------------------------------
// Kernel 2: G = V K^T, split-K=4 over HW=4096 with f32 atomics (G pre-zeroed).
// Both operands are contiguous in GEMM-K (the n axis): pure b64 fills.
// ---------------------------------------------------------------------------
__global__ __launch_bounds__(256) void gram_kernel(
    const __bf16* __restrict__ V, const __bf16* __restrict__ Km,
    float* __restrict__ G) {
  __shared__ __align__(16) __bf16 sV[2][TILE64_];
  __shared__ __align__(16) __bf16 sK[2][TILE128_];

  int d0 = blockIdx.x * 128;         // N (columns of G)
  int c0 = blockIdx.y * 64;          // M (rows of G)
  int ks = blockIdx.z & 3;           // K-split chunk
  int b  = blockIdx.z >> 2;
  const __bf16* Vb = V  + (size_t)b * CI_ * HW_ + (size_t)c0 * HW_;
  const __bf16* Kb = Km + (size_t)b * CI_ * HW_ + (size_t)d0 * HW_;

  int wave = threadIdx.x >> 5;
  int mw = (wave >> 2) * 32;
  int nw = (wave & 3) * 32;

  Acc4 acc = {};
  int kbeg = ks * (HW_ / 4);
  const int NT = (HW_ / 4) / 32;  // 32

  fill_rows_bf<64>(sV[0], Vb + kbeg, HW_);
  fill_rows_bf<128>(sK[0], Kb + kbeg, HW_);

  for (int kt = 0; kt < NT; ++kt) {
    __syncthreads();
    int cur = kt & 1;
    if (kt + 1 < NT) {
      int k0 = kbeg + (kt + 1) * 32;
      fill_rows_bf<64>(sV[cur ^ 1], Vb + k0, HW_);
      fill_rows_bf<128>(sK[cur ^ 1], Kb + k0, HW_);  // K tile is already [d][k]
    }
    mma_step(acc, sV[cur], sK[cur], mw, nw);
  }

  float* Gb = G + (size_t)b * CI_ * CI_;
  atomic_acc_f32(Gb, acc.a00, c0 + mw, d0 + nw);
  atomic_acc_f32(Gb, acc.a01, c0 + mw, d0 + nw + 16);
  atomic_acc_f32(Gb, acc.a10, c0 + mw + 16, d0 + nw);
  atomic_acc_f32(Gb, acc.a11, c0 + mw + 16, d0 + nw + 16);
}

// ---------------------------------------------------------------------------
// Kernel 2b: S = sigmoid(G) -> bf16
// ---------------------------------------------------------------------------
__global__ __launch_bounds__(256) void sigmoid_kernel(const float* __restrict__ G,
                                                      __bf16* __restrict__ S,
                                                      int n) {
  int i = blockIdx.x * 256 + threadIdx.x;
  if (i < n) S[i] = (__bf16)(1.0f / (1.0f + __expf(-G[i])));
}

// ---------------------------------------------------------------------------
// Kernel 3: A = S Q * scale   (M=128, N=4096, K=128 per batch)
// ---------------------------------------------------------------------------
__global__ __launch_bounds__(256) void attn_kernel(const __bf16* __restrict__ S,
                                                   const __bf16* __restrict__ Q,
                                                   __bf16* __restrict__ A,
                                                   float scale) {
  __shared__ __align__(16) __bf16 sS[2][TILE64_];
  __shared__ __align__(16) __bf16 sQ[2][TILE128_];

  int n0 = blockIdx.x * 128;
  int m0 = blockIdx.y * 64;
  int b  = blockIdx.z;
  const __bf16* Sb = S + (size_t)b * CI_ * CI_ + (size_t)m0 * CI_;
  const __bf16* Qb = Q + (size_t)b * CI_ * HW_ + n0;

  int wave = threadIdx.x >> 5;
  int mw = (wave >> 2) * 32;
  int nw = (wave & 3) * 32;

  Acc4 acc = {};
  const int NT = CI_ / 32;  // 4

  fill_rows_bf<64>(sS[0], Sb, CI_);
  fill_BT_bf<128>(sQ[0], Qb, HW_);

  for (int kt = 0; kt < NT; ++kt) {
    __syncthreads();
    int cur = kt & 1;
    if (kt + 1 < NT) {
      int k0 = (kt + 1) * 32;
      fill_rows_bf<64>(sS[cur ^ 1], Sb + k0, CI_);
      fill_BT_bf<128>(sQ[cur ^ 1], Qb + (size_t)k0 * HW_, HW_);
    }
    mma_step(acc, sS[cur], sQ[cur], mw, nw);
  }

  __bf16* Ab = A + (size_t)b * CI_ * HW_;
  store_scaled_bf(Ab, acc.a00, m0 + mw, n0 + nw, scale);
  store_scaled_bf(Ab, acc.a01, m0 + mw, n0 + nw + 16, scale);
  store_scaled_bf(Ab, acc.a10, m0 + mw + 16, n0 + nw, scale);
  store_scaled_bf(Ab, acc.a11, m0 + mw + 16, n0 + nw + 16, scale);
}

// ---------------------------------------------------------------------------
// Kernel 4: out = Wo A + bo + x   (M=256, N=4096, K=128 per batch), f32 out.
// ---------------------------------------------------------------------------
__global__ __launch_bounds__(256) void out_kernel(const float* __restrict__ Wo,
                                                  const float* __restrict__ bo,
                                                  const __bf16* __restrict__ A,
                                                  const float* __restrict__ x,
                                                  float* __restrict__ out) {
  __shared__ __align__(16) __bf16 sW[2][TILE64_];
  __shared__ __align__(16) __bf16 sA[2][TILE128_];

  int n0 = blockIdx.x * 128;
  int m0 = blockIdx.y * 64;   // over CIN_=256
  int b  = blockIdx.z;
  const float*  Wb = Wo + (size_t)m0 * CI_;
  const __bf16* Ab = A + (size_t)b * CI_ * HW_ + n0;

  int wave = threadIdx.x >> 5;
  int mw = (wave >> 2) * 32;
  int nw = (wave & 3) * 32;

  Acc4 acc = {};
  const int NT = CI_ / 32;  // 4

  fill_rows_f32<64>(sW[0], Wb, CI_);
  fill_BT_bf<128>(sA[0], Ab, HW_);

  for (int kt = 0; kt < NT; ++kt) {
    __syncthreads();
    int cur = kt & 1;
    if (kt + 1 < NT) {
      int k0 = (kt + 1) * 32;
      fill_rows_f32<64>(sW[cur ^ 1], Wb + k0, CI_);
      fill_BT_bf<128>(sA[cur ^ 1], Ab + (size_t)k0 * HW_, HW_);
    }
    mma_step(acc, sW[cur], sA[cur], mw, nw);
  }

  // epilogue: + bias + residual, f32 store
  int lane = threadIdx.x & 31;
  size_t base = (size_t)b * CIN_ * HW_;
#pragma unroll
  for (int t = 0; t < 4; ++t) {
    v8f a = (t == 0) ? acc.a00 : (t == 1) ? acc.a01 : (t == 2) ? acc.a10 : acc.a11;
    int n  = n0 + nw + (t & 1) * 16 + (lane & 15);
    int mb = m0 + mw + (t >> 1) * 16 + ((lane >> 4) << 3);
#pragma unroll
    for (int r = 0; r < 8; ++r) {
      size_t idx = base + (size_t)(mb + r) * HW_ + n;
      out[idx] = a[r] + bo[mb + r] + x[idx];
    }
  }
}

// ---------------------------------------------------------------------------
// Launch
// ---------------------------------------------------------------------------
extern "C" void kernel_launch(void* const* d_in, const int* in_sizes, int n_in,
                              void* d_out, int out_size, void* d_ws, size_t ws_size,
                              hipStream_t stream) {
  (void)in_sizes; (void)n_in; (void)out_size; (void)ws_size;
  const float* x  = (const float*)d_in[0];
  const float* Wv = (const float*)d_in[1];
  const float* bv = (const float*)d_in[2];
  const float* Wk = (const float*)d_in[3];
  const float* bk = (const float*)d_in[4];
  const float* Wq = (const float*)d_in[5];
  const float* bq = (const float*)d_in[6];
  const float* Wo = (const float*)d_in[7];
  const float* bo = (const float*)d_in[8];
  float* out = (float*)d_out;

  const size_t per = (size_t)B_ * CI_ * HW_;       // 16M elements
  __bf16* Vbf = (__bf16*)d_ws;
  __bf16* Kbf = Vbf + per;
  __bf16* Qbf = Kbf + per;
  __bf16* Abf = Qbf + per;
  float*  Gf  = (float*)(Abf + per);
  __bf16* Sbf = (__bf16*)(Gf + (size_t)B_ * CI_ * CI_);

  // zero G accumulator for split-K atomics (graph-capture safe)
  hipMemsetAsync(Gf, 0, (size_t)B_ * CI_ * CI_ * sizeof(float), stream);

  qkv_kernel<<<dim3(HW_ / 64, CI_ / 64, B_), 256, 0, stream>>>(
      x, Wv, bv, Wk, bk, Wq, bq, Vbf, Kbf, Qbf);

  gram_kernel<<<dim3(CI_ / 128, CI_ / 64, B_ * 4), 256, 0, stream>>>(Vbf, Kbf, Gf);

  int ng = B_ * CI_ * CI_;
  sigmoid_kernel<<<(ng + 255) / 256, 256, 0, stream>>>(Gf, Sbf, ng);

  const float scale = 0.0013810679320049757f;  // 1/sqrt(CI*HW) = 1/sqrt(524288)
  attn_kernel<<<dim3(HW_ / 128, CI_ / 64, B_), 256, 0, stream>>>(Sbf, Qbf, Abf, scale);

  out_kernel<<<dim3(HW_ / 128, CIN_ / 64, B_), 256, 0, stream>>>(Wo, bo, Abf, x, out);
}
